// CausalSelfAttention_35905926595323
// MI455X (gfx1250) — matrix operand order repaired and checked by reading, not compiled, not run on hardware
//
#include <hip/hip_runtime.h>
#include <cstdint>
#include <cmath>

// ---------------- problem constants ----------------
#define DIMN 1024
#define NH   16
#define NKV  4
#define HD   64
#define GRP  4          // NH / NKV
#define BB   2
#define TSEQ 2048
#define NT   (BB*TSEQ)  // 4096 tokens
#define EPSF 1.1920928955078125e-07f
#define ATT_SCALE 0.125f  // 1/sqrt(64)

typedef __attribute__((ext_vector_type(16))) _Float16 v16h;
typedef __attribute__((ext_vector_type(8)))  _Float16 v8h;
typedef __attribute__((ext_vector_type(8)))  float    v8f;
typedef __attribute__((ext_vector_type(4)))  int      v4i;

// ---------------- CDNA5 async-to-LDS path (guarded) ----------------
#if defined(__gfx1250__) && __has_builtin(__builtin_amdgcn_global_load_async_to_lds_b128)
#define HAS_ASYNC 1
#else
#define HAS_ASYNC 0
#endif

__device__ __forceinline__ void stage16B(const _Float16* g, _Float16* l) {
#if HAS_ASYNC
  __builtin_amdgcn_global_load_async_to_lds_b128(
      (__attribute__((address_space(1))) v4i*)(uintptr_t)g,
      (__attribute__((address_space(3))) v4i*)l, 0, 0);
#else
  *(v8h*)l = *(const v8h*)g;
#endif
}
__device__ __forceinline__ void stage32B(const _Float16* g, _Float16* l) {
  stage16B(g, l);
  stage16B(g + 8, l + 8);
}
__device__ __forceinline__ void wait_async_lds() {
#if HAS_ASYNC
#if __has_builtin(__builtin_amdgcn_s_wait_asynccnt)
  __builtin_amdgcn_s_wait_asynccnt(0);
#else
  asm volatile("s_wait_asynccnt 0" ::: "memory");
#endif
#endif
}

// ---------------- wave32 xor-shuffle reductions via ds_swizzle ----------------
template <int XM>
__device__ __forceinline__ float swz_xor(float v) {
  int i = __builtin_amdgcn_ds_swizzle(__builtin_bit_cast(int, v), (XM << 10) | 0x1F);
  return __builtin_bit_cast(float, i);
}
__device__ __forceinline__ float wave_sum32(float v) {
  v += swz_xor<16>(v); v += swz_xor<8>(v); v += swz_xor<4>(v);
  v += swz_xor<2>(v);  v += swz_xor<1>(v);
  return v;
}
__device__ __forceinline__ float half_sum16(float v) {
  v += swz_xor<8>(v); v += swz_xor<4>(v); v += swz_xor<2>(v); v += swz_xor<1>(v);
  return v;
}
__device__ __forceinline__ float half_max16(float v) {
  v = fmaxf(v, swz_xor<8>(v)); v = fmaxf(v, swz_xor<4>(v));
  v = fmaxf(v, swz_xor<2>(v)); v = fmaxf(v, swz_xor<1>(v));
  return v;
}

// ---------------- WMMA helpers ----------------
// A-matrix 16x32 f16 layout: lanes rows M=lane%16 hold K=16g..16g+15 (contiguous, matching B)
__device__ __forceinline__ v16h load_a_frag(const _Float16* rowp, int grp) {
  v8h lo = *(const v8h*)(rowp + 16 * grp);
  v8h hi = *(const v8h*)(rowp + 16 * grp + 8);
  v16h r;
#pragma unroll
  for (int i = 0; i < 8; ++i) { r[i] = lo[i]; r[8 + i] = hi[i]; }
  return r;
}
__device__ __forceinline__ v8f wmma_f16(v16h a, v16h b, v8f c) {
  return __builtin_amdgcn_wmma_f32_16x16x32_f16(false, a, false, b, (short)0, c,
                                                false, false);
}

// ---------------- elementwise f32 -> f16 cast ----------------
__global__ void cast_f32_f16_kernel(const float* __restrict__ in,
                                    _Float16* __restrict__ out, int n) {
  int i = blockIdx.x * blockDim.x + threadIdx.x;
  if (i < n) out[i] = (_Float16)in[i];
}

// ---------------- row RMS-norm of x, emit f16 x and f16 rmsnorm(x) ----------------
__global__ __launch_bounds__(256) void rmsnorm_cast_kernel(
    const float* __restrict__ x, _Float16* __restrict__ xh,
    _Float16* __restrict__ xqkh) {
  __shared__ float red[256];
  const int row = blockIdx.x;
  const float* xr = x + (size_t)row * DIMN;
  float ss = 0.f;
  for (int i = threadIdx.x; i < DIMN; i += 256) { float v = xr[i]; ss += v * v; }
  red[threadIdx.x] = ss;
  __syncthreads();
  for (int s = 128; s > 0; s >>= 1) {
    if (threadIdx.x < s) red[threadIdx.x] += red[threadIdx.x + s];
    __syncthreads();
  }
  const float r = rsqrtf(red[0] / (float)DIMN + EPSF);
  for (int i = threadIdx.x; i < DIMN; i += 256) {
    float v = xr[i];
    xh[(size_t)row * DIMN + i] = (_Float16)v;
    xqkh[(size_t)row * DIMN + i] = (_Float16)(v * r);
  }
}

// ---------------- generic WMMA GEMM: C[M,N] = A[M,K] * W[N,K]^T ----------------
// 128x128 block, BK=32, 8 waves (4x2), each wave 32x64 (2x4 16x16 tiles).
#define GBM 128
#define GBN 128
#define GBK 32
__global__ __launch_bounds__(256) void gemm_wmma_kernel(
    const _Float16* __restrict__ A, const _Float16* __restrict__ W,
    float* __restrict__ C, int M, int N, int K) {
  __shared__ _Float16 As[GBM][GBK] __attribute__((aligned(64)));
  __shared__ _Float16 Bs[GBN][GBK] __attribute__((aligned(64)));

  const int tid = threadIdx.x;
  const int lane = tid & 31, wave = tid >> 5;
  const int wr = wave >> 1, wc = wave & 1;
  const int grp = lane >> 4, mr = lane & 15;
  const int m0 = blockIdx.y * GBM;
  const int n0 = blockIdx.x * GBN;
  const int crow = tid >> 1, cseg = tid & 1;

  v8f acc[2][4];
#pragma unroll
  for (int mt = 0; mt < 2; ++mt)
#pragma unroll
    for (int nt = 0; nt < 4; ++nt) { v8f z = {}; acc[mt][nt] = z; }

  for (int k0 = 0; k0 < K; k0 += GBK) {
    stage32B(A + (size_t)(m0 + crow) * K + k0 + cseg * 16, &As[crow][cseg * 16]);
    stage32B(W + (size_t)(n0 + crow) * K + k0 + cseg * 16, &Bs[crow][cseg * 16]);
    wait_async_lds();
    __syncthreads();

    v16h afr[2];
#pragma unroll
    for (int mt = 0; mt < 2; ++mt)
      afr[mt] = load_a_frag(&As[wr * 32 + mt * 16 + mr][0], grp);
#pragma unroll
    for (int nt = 0; nt < 4; ++nt) {
      // B 32x16: lane holds column n = lane%16, 16 K-contiguous halves
      v16h bfr = *(const v16h*)&Bs[wc * 64 + nt * 16 + mr][grp * 16];
#pragma unroll
      for (int mt = 0; mt < 2; ++mt)
        acc[mt][nt] = wmma_f16(afr[mt], bfr, acc[mt][nt]);
    }
    __syncthreads();
  }

#pragma unroll
  for (int mt = 0; mt < 2; ++mt)
#pragma unroll
    for (int nt = 0; nt < 4; ++nt)
#pragma unroll
      for (int r = 0; r < 8; ++r) {
        int row = m0 + wr * 32 + mt * 16 + r + 8 * grp;  // C/D: M = r + 8*(lane/16)
        int col = n0 + wc * 64 + nt * 16 + mr;           //      N = lane%16
        C[(size_t)row * N + col] = acc[mt][nt][r];
      }
}

// ---------------- per-head RMS-norm + RoPE (+ gain), f32 in -> f16 out ----------------
__global__ __launch_bounds__(128) void qknorm_rope_kernel(
    const float* __restrict__ in, _Float16* __restrict__ out,
    const float* __restrict__ gain, int nh) {
  const int wid = (blockIdx.x * 128 + threadIdx.x) >> 5;
  const int lane = threadIdx.x & 31;
  if (wid >= NT * nh) return;
  const int t = wid / nh, h = wid % nh;
  const int ld = nh * HD;
  const float* p = in + (size_t)t * ld + h * HD;
  float q1 = p[lane], q2 = p[lane + 32];
  float ss = wave_sum32(q1 * q1 + q2 * q2);
  float r = rsqrtf(ss / (float)HD + EPSF);
  q1 *= r; q2 *= r;
  const int pos = t % TSEQ;
  float inv = __powf(10000.0f, -(float)lane / 32.0f);
  float fr = (float)pos * inv;
  float s, c;
  __sincosf(fr, &s, &c);
  float o1 = q1 * c + q2 * s;
  float o2 = -q1 * s + q2 * c;
  if (gain) { float g = gain[h]; o1 *= g; o2 *= g; }
  _Float16* op = out + (size_t)t * ld + h * HD;
  op[lane] = (_Float16)o1;
  op[lane + 32] = (_Float16)o2;
}

// ---------------- flash attention (causal, GQA), WMMA QK^T and PV ----------------
__global__ __launch_bounds__(128) void attn_kernel(const _Float16* __restrict__ Qh,
                                                   const _Float16* __restrict__ Kh,
                                                   const _Float16* __restrict__ Vh,
                                                   float* __restrict__ Y) {
  __shared__ _Float16 ldsK[64][HD] __attribute__((aligned(64)));   // [key][hd]
  __shared__ _Float16 ldsVt[HD][64] __attribute__((aligned(64)));  // [hd][key]
  __shared__ _Float16 ldsP[4][16][64] __attribute__((aligned(64)));

  const int tid = threadIdx.x;
  const int lane = tid & 31, wave = tid >> 5;
  const int grp = lane >> 4, mr = lane & 15;

  const int qb = blockIdx.x;         // T/64 query blocks
  const int bh = blockIdx.y;         // B*H
  const int b = bh / NH, h = bh % NH;
  const int g = h / GRP;
  const int tbase = b * TSEQ;
  const int q0 = qb * 64 + wave * 16;
  const int q_last = q0 + 15;

  // Q fragments (A layout), two 32-wide K chunks over HD=64
  const _Float16* qrow = Qh + (size_t)(tbase + q0 + mr) * (NH * HD) + h * HD;
  v16h qa[2];
#pragma unroll
  for (int c = 0; c < 2; ++c) qa[c] = load_a_frag(qrow + c * 32, grp);

  v8f oacc[4];
#pragma unroll
  for (int ot = 0; ot < 4; ++ot) { v8f z = {}; oacc[ot] = z; }
  float mrun[8], lrun[8];
#pragma unroll
  for (int r = 0; r < 8; ++r) { mrun[r] = -INFINITY; lrun[r] = 0.f; }

  const int crow = tid >> 1, cseg = tid & 1;
  const int nkb = qb + 1;

  for (int kb = 0; kb < nkb; ++kb) {
    const int k0 = kb * 64;
    // cooperative stage: K tile async; V tile transposed
    {
      const _Float16* ksrc =
          Kh + (size_t)(tbase + k0 + crow) * (NKV * HD) + g * HD + cseg * 32;
      stage32B(ksrc, &ldsK[crow][cseg * 32]);
      const _Float16* vsrc =
          Vh + (size_t)(tbase + k0 + crow) * (NKV * HD) + g * HD + cseg * 32;
      v8h vv[4];
#pragma unroll
      for (int q = 0; q < 4; ++q) vv[q] = *(const v8h*)(vsrc + q * 8);
#pragma unroll
      for (int j = 0; j < 32; ++j) ldsVt[cseg * 32 + j][crow] = vv[j >> 3][j & 7];
    }
    wait_async_lds();
    __syncthreads();

    if (k0 <= q_last) {  // wave-uniform causal skip
      // S = Q K^T
      v8f sacc[4];
#pragma unroll
      for (int nt = 0; nt < 4; ++nt) {
        v8f z = {};
        sacc[nt] = z;
#pragma unroll
        for (int c = 0; c < 2; ++c) {
          v16h bfr = *(const v16h*)&ldsK[nt * 16 + mr][c * 32 + grp * 16];
          sacc[nt] = wmma_f16(qa[c], bfr, sacc[nt]);
        }
      }
      // scale + causal mask (element (r,lane): q=q0+r+8*grp, k=k0+nt*16+mr)
#pragma unroll
      for (int nt = 0; nt < 4; ++nt) {
        const int kpos = k0 + nt * 16 + mr;
#pragma unroll
        for (int r = 0; r < 8; ++r) {
          const int qpos = q0 + r + 8 * grp;
          float s = sacc[nt][r] * ATT_SCALE;
          sacc[nt][r] = (kpos <= qpos) ? s : -INFINITY;
        }
      }
      // online softmax (row spans 16 lanes of this lane-group)
      float alpha[8];
#pragma unroll
      for (int r = 0; r < 8; ++r) {
        float v = fmaxf(fmaxf(sacc[0][r], sacc[1][r]),
                        fmaxf(sacc[2][r], sacc[3][r]));
        v = half_max16(v);
        float mn = fmaxf(mrun[r], v);
        alpha[r] = __expf(mrun[r] - mn);
        mrun[r] = mn;
      }
      float ladd[8];
#pragma unroll
      for (int r = 0; r < 8; ++r) ladd[r] = 0.f;
#pragma unroll
      for (int nt = 0; nt < 4; ++nt)
#pragma unroll
        for (int r = 0; r < 8; ++r) {
          float pv = __expf(sacc[nt][r] - mrun[r]);
          ladd[r] += pv;
          ldsP[wave][r + 8 * grp][nt * 16 + mr] = (_Float16)pv;
        }
#pragma unroll
      for (int r = 0; r < 8; ++r)
        lrun[r] = lrun[r] * alpha[r] + half_sum16(ladd[r]);
#pragma unroll
      for (int ot = 0; ot < 4; ++ot)
#pragma unroll
        for (int r = 0; r < 8; ++r) oacc[ot][r] *= alpha[r];
      // O += P @ V
#pragma unroll
      for (int c = 0; c < 2; ++c) {
        v16h pa = load_a_frag(&ldsP[wave][mr][c * 32], grp);
#pragma unroll
        for (int ot = 0; ot < 4; ++ot) {
          v16h bv = *(const v16h*)&ldsVt[ot * 16 + mr][c * 32 + grp * 16];
          oacc[ot] = wmma_f16(pa, bv, oacc[ot]);
        }
      }
    }
    __syncthreads();
  }

#pragma unroll
  for (int ot = 0; ot < 4; ++ot)
#pragma unroll
    for (int r = 0; r < 8; ++r) {
      float y = oacc[ot][r] / lrun[r];
      Y[(size_t)(tbase + q0 + r + 8 * grp) * DIMN + h * HD + ot * 16 + mr] = y;
    }
}

// ---------------- v-direction rejection, write f16 Y ----------------
__global__ __launch_bounds__(128) void vproj_kernel(const float* __restrict__ Y,
                                                    const float* __restrict__ Vf,
                                                    _Float16* __restrict__ Yh) {
  const int wid = (blockIdx.x * 128 + threadIdx.x) >> 5;
  const int lane = threadIdx.x & 31;
  if (wid >= NT * NKV) return;
  const int t = wid / NKV, g = wid % NKV;
  const float* vp = Vf + (size_t)t * (NKV * HD) + g * HD;
  float v1 = vp[lane], v2 = vp[lane + 32];
  float nrm = sqrtf(wave_sum32(v1 * v1 + v2 * v2));
  float inv = 1.0f / fmaxf(nrm, 1e-12f);
  float n1 = v1 * inv, n2 = v2 * inv;
#pragma unroll
  for (int j = 0; j < GRP; ++j) {
    const int h = g * GRP + j;
    const float* yp = Y + (size_t)t * DIMN + h * HD;
    float y1 = yp[lane], y2 = yp[lane + 32];
    float d = wave_sum32(y1 * n1 + y2 * n2);
    y1 -= d * n1;
    y2 -= d * n2;
    _Float16* op = Yh + (size_t)t * DIMN + h * HD;
    op[lane] = (_Float16)y1;
    op[lane + 32] = (_Float16)y2;
  }
}

// ---------------- host orchestration ----------------
extern "C" void kernel_launch(void* const* d_in, const int* in_sizes, int n_in,
                              void* d_out, int out_size, void* d_ws, size_t ws_size,
                              hipStream_t stream) {
  (void)in_sizes; (void)n_in; (void)out_size; (void)ws_size;
  const float* x   = (const float*)d_in[0];
  const float* Wq  = (const float*)d_in[1];
  const float* Wk  = (const float*)d_in[2];
  const float* Wv  = (const float*)d_in[3];
  const float* Wp  = (const float*)d_in[4];
  const float* qg  = (const float*)d_in[5];
  float* out = (float*)d_out;

  char* p = (char*)d_ws;
  auto alloc = [&](size_t bytes) {
    char* r = p;
    p += (bytes + 255) & ~(size_t)255;
    return r;
  };
  _Float16* xh   = (_Float16*)alloc((size_t)NT * DIMN * 2);
  _Float16* xqkh = (_Float16*)alloc((size_t)NT * DIMN * 2);
  _Float16* Wqh  = (_Float16*)alloc((size_t)DIMN * DIMN * 2);
  _Float16* Wkh  = (_Float16*)alloc((size_t)NKV * HD * DIMN * 2);
  _Float16* Wvh  = (_Float16*)alloc((size_t)NKV * HD * DIMN * 2);
  _Float16* Wph  = (_Float16*)alloc((size_t)DIMN * DIMN * 2);
  float*    Qf   = (float*)alloc((size_t)NT * DIMN * 4);
  float*    Kf   = (float*)alloc((size_t)NT * NKV * HD * 4);
  float*    Vf   = (float*)alloc((size_t)NT * NKV * HD * 4);
  _Float16* Qhh  = (_Float16*)alloc((size_t)NT * DIMN * 2);
  _Float16* Khh  = (_Float16*)alloc((size_t)NT * NKV * HD * 2);
  _Float16* Vhh  = (_Float16*)alloc((size_t)NT * NKV * HD * 2);
  float*    Yf   = (float*)alloc((size_t)NT * DIMN * 4);
  _Float16* Yh   = (_Float16*)alloc((size_t)NT * DIMN * 2);

  // 1) weight casts
  cast_f32_f16_kernel<<<(DIMN * DIMN + 255) / 256, 256, 0, stream>>>(Wq, Wqh, DIMN * DIMN);
  cast_f32_f16_kernel<<<(NKV * HD * DIMN + 255) / 256, 256, 0, stream>>>(Wk, Wkh, NKV * HD * DIMN);
  cast_f32_f16_kernel<<<(NKV * HD * DIMN + 255) / 256, 256, 0, stream>>>(Wv, Wvh, NKV * HD * DIMN);
  cast_f32_f16_kernel<<<(DIMN * DIMN + 255) / 256, 256, 0, stream>>>(Wp, Wph, DIMN * DIMN);
  // 2) rms-norm of x + f16 casts
  rmsnorm_cast_kernel<<<NT, 256, 0, stream>>>(x, xh, xqkh);
  // 3) QKV projections (WMMA)
  gemm_wmma_kernel<<<dim3(DIMN / GBN, NT / GBM), 256, 0, stream>>>(xqkh, Wqh, Qf, NT, DIMN, DIMN);
  gemm_wmma_kernel<<<dim3(NKV * HD / GBN, NT / GBM), 256, 0, stream>>>(xqkh, Wkh, Kf, NT, NKV * HD, DIMN);
  gemm_wmma_kernel<<<dim3(NKV * HD / GBN, NT / GBM), 256, 0, stream>>>(xh, Wvh, Vf, NT, NKV * HD, DIMN);
  // 4) per-head norm + RoPE (+ q gain)
  qknorm_rope_kernel<<<(NT * NH) / 4, 128, 0, stream>>>(Qf, Qhh, qg, NH);
  qknorm_rope_kernel<<<(NT * NKV) / 4, 128, 0, stream>>>(Kf, Khh, nullptr, NKV);
  cast_f32_f16_kernel<<<(NT * NKV * HD + 255) / 256, 256, 0, stream>>>(Vf, Vhh, NT * NKV * HD);
  // 5) causal flash attention
  attn_kernel<<<dim3(TSEQ / 64, BB * NH), 128, 0, stream>>>(Qhh, Khh, Vhh, Yf);
  // 6) v-direction rejection
  vproj_kernel<<<(NT * NKV) / 4, 128, 0, stream>>>(Yf, Vf, Yh);
  // 7) output projection (WMMA) -> d_out (f32)
  gemm_wmma_kernel<<<dim3(DIMN / GBN, NT / GBM), 256, 0, stream>>>(Yh, Wph, out, NT, DIMN, DIMN);
}
